// APPNP_29506425323819
// MI455X (gfx1250) — matrix operand order repaired and checked by reading, not compiled, not run on hardware
//
#include <hip/hip_runtime.h>
#include <hip/hip_bf16.h>

// APPNP (k=1) for MI455X / gfx1250.
// Pipeline: zero -> degrees (atomics) -> norms -> W fp32->bf16 -> WMMA GEMM (h0)
//           -> edge scatter (gather*norm_src, atomic add into d_out)
//           -> finalize (out = 0.8*out*norm_dst + 0.2*h0).
// Assumes F % 32 == 0, C == 64, N % 32 == 0 (true for the reference shapes:
// N=100000, F=512, C=64, E=3200000).

typedef __attribute__((ext_vector_type(16))) __bf16 v16bf;
typedef __attribute__((ext_vector_type(8)))  float  v8f;

#define ALPHA 0.2f

// ---------------------------------------------------------------- zero init
__global__ void appnp_zero_kernel(float* __restrict__ out, float* __restrict__ dsrc,
                                  float* __restrict__ ddst, int NC, int N) {
    int i = blockIdx.x * blockDim.x + threadIdx.x;
    if (i < NC) out[i] = 0.0f;
    if (i < N) { dsrc[i] = 0.0f; ddst[i] = 0.0f; }
}

// ---------------------------------------------------------------- degrees
__global__ void appnp_degree_kernel(const int* __restrict__ src, const int* __restrict__ dst,
                                    float* __restrict__ dsrc, float* __restrict__ ddst, int E) {
    int e = blockIdx.x * blockDim.x + threadIdx.x;
    if (e < E) {
        atomicAdd(&dsrc[src[e]], 1.0f);   // out-degree
        atomicAdd(&ddst[dst[e]], 1.0f);   // in-degree
    }
}

// ------------------------------------------------------ deg -> rsqrt(clamp)
__global__ void appnp_norm_kernel(float* __restrict__ dsrc, float* __restrict__ ddst, int N) {
    int i = blockIdx.x * blockDim.x + threadIdx.x;
    if (i < N) {
        dsrc[i] = rsqrtf(fmaxf(dsrc[i], 1.0f));
        ddst[i] = rsqrtf(fmaxf(ddst[i], 1.0f));
    }
}

// ---------------------------------------------------------- W fp32 -> bf16
__global__ void appnp_cvt_w_kernel(const float* __restrict__ W, __bf16* __restrict__ Wbf, int n) {
    int i = blockIdx.x * blockDim.x + threadIdx.x;
    if (i < n) Wbf[i] = (__bf16)W[i];
}

// ------------------------------------------------- pack 16 fp32 -> A frag
__device__ __forceinline__ v16bf appnp_pack_a(const float* __restrict__ p) {
    const float4 x0 = *(const float4*)(p);
    const float4 x1 = *(const float4*)(p + 4);
    const float4 y0 = *(const float4*)(p + 8);
    const float4 y1 = *(const float4*)(p + 12);
    v16bf a;
    a[0]  = (__bf16)x0.x; a[1]  = (__bf16)x0.y; a[2]  = (__bf16)x0.z; a[3]  = (__bf16)x0.w;
    a[4]  = (__bf16)x1.x; a[5]  = (__bf16)x1.y; a[6]  = (__bf16)x1.z; a[7]  = (__bf16)x1.w;
    a[8]  = (__bf16)y0.x; a[9]  = (__bf16)y0.y; a[10] = (__bf16)y0.z; a[11] = (__bf16)y0.w;
    a[12] = (__bf16)y1.x; a[13] = (__bf16)y1.y; a[14] = (__bf16)y1.z; a[15] = (__bf16)y1.w;
    return a;
}

// ------------------------------------------------------------- WMMA GEMM
// h0[M, 64] = in_feat[M, F] @ W[64, F]^T + b[64]
// One wave -> 32 rows x 64 cols (2 M-tiles x 4 N-tiles = 8 accumulators);
// each B fragment loaded once per K-step feeds two WMMAs (halves W traffic).
__global__ void __launch_bounds__(256)
appnp_gemm_wmma_kernel(const float* __restrict__ A, const __bf16* __restrict__ Wbf,
                       const float* __restrict__ bias, float* __restrict__ h0,
                       int mtile2, int F) {
    const int lane = threadIdx.x & 31;
    const int wave = blockIdx.x * (blockDim.x >> 5) + (threadIdx.x >> 5);
    if (wave >= mtile2) return;            // whole-wave guard: EXEC all-1s inside

    const int r0   = wave << 5;            // 32 rows per wave
    const int half = lane >> 4;            // 0: lanes 0-15, 1: lanes 16-31
    const int l16  = lane & 15;

    // C/D layout: element j -> row base + j + 8*half, col t*16 + l16 (fixed/lane)
    v8f acc[2][4];
#pragma unroll
    for (int t = 0; t < 4; ++t) {
        const float bv = bias[t * 16 + l16];
#pragma unroll
        for (int j = 0; j < 8; ++j) { acc[0][t][j] = bv; acc[1][t][j] = bv; }
    }

    // A fragment rows for this lane (two M-tiles), bf16 16x32 layout:
    //   elems 0-15 = K [k0 + 16*half .. +15]
    const float* ap0 = A + (size_t)(r0 + l16) * (size_t)F + half * 16;
    const float* ap1 = A + (size_t)(r0 + 16 + l16) * (size_t)F + half * 16;

    // B fragment (32x16 bf16): lane col = t*16 + l16, K = k0 + 16*half .. +15
    // -> 16 contiguous bf16 in row-major W
    const __bf16* bp0 = Wbf + (size_t)(l16)      * (size_t)F + half * 16;
    const __bf16* bp1 = Wbf + (size_t)(16 + l16) * (size_t)F + half * 16;
    const __bf16* bp2 = Wbf + (size_t)(32 + l16) * (size_t)F + half * 16;
    const __bf16* bp3 = Wbf + (size_t)(48 + l16) * (size_t)F + half * 16;

    for (int k0 = 0; k0 < F; k0 += 32) {
        const v16bf a0 = appnp_pack_a(ap0 + k0);
        const v16bf a1 = appnp_pack_a(ap1 + k0);
        const v16bf b0 = *(const v16bf*)(bp0 + k0);
        const v16bf b1 = *(const v16bf*)(bp1 + k0);
        const v16bf b2 = *(const v16bf*)(bp2 + k0);
        const v16bf b3 = *(const v16bf*)(bp3 + k0);

        acc[0][0] = __builtin_amdgcn_wmma_f32_16x16x32_bf16(false, a0, false, b0, (short)0, acc[0][0], false, false);
        acc[1][0] = __builtin_amdgcn_wmma_f32_16x16x32_bf16(false, a1, false, b0, (short)0, acc[1][0], false, false);
        acc[0][1] = __builtin_amdgcn_wmma_f32_16x16x32_bf16(false, a0, false, b1, (short)0, acc[0][1], false, false);
        acc[1][1] = __builtin_amdgcn_wmma_f32_16x16x32_bf16(false, a1, false, b1, (short)0, acc[1][1], false, false);
        acc[0][2] = __builtin_amdgcn_wmma_f32_16x16x32_bf16(false, a0, false, b2, (short)0, acc[0][2], false, false);
        acc[1][2] = __builtin_amdgcn_wmma_f32_16x16x32_bf16(false, a1, false, b2, (short)0, acc[1][2], false, false);
        acc[0][3] = __builtin_amdgcn_wmma_f32_16x16x32_bf16(false, a0, false, b3, (short)0, acc[0][3], false, false);
        acc[1][3] = __builtin_amdgcn_wmma_f32_16x16x32_bf16(false, a1, false, b3, (short)0, acc[1][3], false, false);
    }

    // store: lane writes column (t*16 + l16), rows r0 + 16*m + j + 8*half
#pragma unroll
    for (int m = 0; m < 2; ++m) {
#pragma unroll
        for (int t = 0; t < 4; ++t) {
            const int col = t * 16 + l16;
            float* hp = h0 + (size_t)(r0 + m * 16 + half * 8) * 64 + col;
#pragma unroll
            for (int j = 0; j < 8; ++j) {
                hp[(size_t)j * 64] = acc[m][t][j];
            }
        }
    }
}

// --------------------------------------------------------- edge scatter
// 64 threads per edge (C == 64): fully coalesced 256B gather + coalesced
// global_atomic_add_f32 into d_out (L2-resident: 25.6 MB << 192 MB).
__global__ void appnp_scatter_kernel(const float* __restrict__ h0,
                                     const float* __restrict__ nsrc,
                                     const int* __restrict__ src, const int* __restrict__ dst,
                                     float* __restrict__ out, int E) {
    const long long t = (long long)blockIdx.x * blockDim.x + threadIdx.x;
    const int e = (int)(t >> 6);          // C == 64
    const int c = (int)(t & 63);
    if (e >= E) return;
    const int s = src[e];
    const int d = dst[e];
    const float v = h0[(size_t)s * 64 + c] * nsrc[s];
    atomicAdd(&out[(size_t)d * 64 + c], v);
}

// ------------------------------------------------------------- finalize
__global__ void appnp_finalize_kernel(float* __restrict__ out, const float* __restrict__ h0,
                                      const float* __restrict__ ndst, int NC) {
    int i = blockIdx.x * blockDim.x + threadIdx.x;
    if (i < NC) {
        const int n = i >> 6;             // C == 64
        out[i] = (1.0f - ALPHA) * out[i] * ndst[n] + ALPHA * h0[i];
    }
}

extern "C" void kernel_launch(void* const* d_in, const int* in_sizes, int n_in,
                              void* d_out, int out_size, void* d_ws, size_t ws_size,
                              hipStream_t stream) {
    const float* in_feat = (const float*)d_in[0];
    const float* W       = (const float*)d_in[1];
    const float* bias    = (const float*)d_in[2];
    const int*   src     = (const int*)d_in[3];
    const int*   dst     = (const int*)d_in[4];
    float*       out     = (float*)d_out;

    const int C = in_sizes[2];            // 64
    const int N = out_size / C;           // 100000
    const int F = in_sizes[0] / N;        // 512
    const int E = in_sizes[3];            // 3200000
    const int NC = N * C;

    // Workspace layout (256B aligned): h0 | norm_src | norm_dst | W_bf16
    char* ws = (char*)d_ws;
    size_t off = 0;
    float* h0 = (float*)(ws + off);
    off += ((size_t)NC * sizeof(float) + 255) & ~(size_t)255;
    float* norm_src = (float*)(ws + off);
    off += ((size_t)N * sizeof(float) + 255) & ~(size_t)255;
    float* norm_dst = (float*)(ws + off);
    off += ((size_t)N * sizeof(float) + 255) & ~(size_t)255;
    __bf16* Wbf = (__bf16*)(ws + off);

    const int TB = 256;

    // 1) zero d_out + degree accumulators
    appnp_zero_kernel<<<(NC + TB - 1) / TB, TB, 0, stream>>>(out, norm_src, norm_dst, NC, N);

    // 2) degrees via fp32 atomics
    appnp_degree_kernel<<<(E + TB - 1) / TB, TB, 0, stream>>>(src, dst, norm_src, norm_dst, E);

    // 3) degrees -> D^{-1/2} in place
    appnp_norm_kernel<<<(N + TB - 1) / TB, TB, 0, stream>>>(norm_src, norm_dst, N);

    // 4) W fp32 -> bf16 (64 KB, stays hot in cache for the GEMM)
    const int CF = C * F;
    appnp_cvt_w_kernel<<<(CF + TB - 1) / TB, TB, 0, stream>>>(W, Wbf, CF);

    // 5) h0 = in_feat @ W^T + b via v_wmma_f32_16x16x32_bf16
    //    one wave = 32 rows; N = 100000 = 3125 * 32 exactly
    const int mtile2 = N / 32;
    const int wavesPerBlock = TB / 32;               // 8
    const int gemmBlocks = (mtile2 + wavesPerBlock - 1) / wavesPerBlock;
    appnp_gemm_wmma_kernel<<<gemmBlocks, TB, 0, stream>>>(in_feat, Wbf, bias, h0, mtile2, F);

    // 6) gather * norm_src -> atomic scatter-add into d_out (64 threads/edge)
    const long long scatterThreads = (long long)E * 64;
    const int scatterBlocks = (int)((scatterThreads + TB - 1) / TB);
    appnp_scatter_kernel<<<scatterBlocks, TB, 0, stream>>>(h0, norm_src, src, dst, out, E);

    // 7) out = (1-alpha) * out * norm_dst + alpha * h0
    appnp_finalize_kernel<<<(NC + TB - 1) / TB, TB, 0, stream>>>(out, h0, norm_dst, NC);
}
